// EdgeGNN_53678501265910
// MI455X (gfx1250) — compile-verified, matrix operand order checked by reading, not run on hardware
//
#include <hip/hip_runtime.h>
#include <cstddef>

typedef __attribute__((ext_vector_type(16))) __bf16 v16bf;
typedef __attribute__((ext_vector_type(8)))  __bf16 v8bf;
typedef __attribute__((ext_vector_type(8)))  float  v8f;

#define WMMA_BF16(a, b, c) \
    __builtin_amdgcn_wmma_f32_16x16x32_bf16(false, (a), false, (b), (short)0, (c), false, false)

constexpr int Bc = 4;
constexpr int Nc = 10000;
constexpr int Ec = 320000;
constexpr int FNc = 16;     // features per node
constexpr int INDIM = 33;   // 2*FN + 1
constexpr int EH  = 64;
constexpr int EH2 = 32;
constexpr int EOUT = 64;
constexpr int NH = 12;

// ---------------------------------------------------------------- zero agg
__global__ void egnn_zero_kernel(float4* p, int n4) {
    int i = blockIdx.x * blockDim.x + threadIdx.x;
    int stride = gridDim.x * blockDim.x;
    float4 z = {0.f, 0.f, 0.f, 0.f};
    for (; i < n4; i += stride) p[i] = z;
}

// ---------------------------------------------------------------- edge MLP
// 8 waves / block; each wave processes 16-edge tiles with WMMA bf16.
__global__ __launch_bounds__(256) void egnn_edge_kernel(
    const float* __restrict__ nf,      // (B,N,FN)
    const float* __restrict__ ew,      // (B,E)
    const int*   __restrict__ esrc,    // (E)
    const int*   __restrict__ etgt,    // (E)
    const float* __restrict__ W1, const float* __restrict__ b1,   // (33,64),(64)
    const float* __restrict__ W2, const float* __restrict__ b2,   // (64,32),(32)
    const float* __restrict__ W3, const float* __restrict__ b3,   // (32,64),(64)
    float* __restrict__ agg)           // (B,N,64) atomically accumulated
{
    // Weight fragments in LDS, pre-laid-out in WMMA B-fragment order:
    // frag(s,t): 32 lanes x 16 bf16, element j of lane L maps to
    //   n = t*16 + (L&15),  k = s*32 + ((j<8 ? j : j+8) + 8*(L>>4))
    __shared__ __attribute__((aligned(32))) __bf16 w1f[2 * 4 * 32 * 16]; // K=64, Nt=4
    __shared__ __attribute__((aligned(32))) __bf16 w2f[2 * 2 * 32 * 16]; // K=64, Nt=2
    __shared__ __attribute__((aligned(32))) __bf16 w3f[1 * 4 * 32 * 16]; // K=32, Nt=4
    __shared__ float b1f[64], b2f[32], b3f[64];
    // Per-wave activation staging (C-layout store -> A-layout load)
    __shared__ __attribute__((aligned(32))) __bf16 h1s[8 * 16 * 72];     // 16 x 64 (+pad)
    __shared__ __attribute__((aligned(32))) __bf16 h2s[8 * 16 * 40];     // 16 x 32 (+pad)
    __shared__ int idxS[8 * 16], idxT[8 * 16];

    const int tidb = threadIdx.x;

    // --- build W1 fragments (zero-pad K 33..63) ---
    for (int q = tidb; q < 4096; q += 256) {
        int j = q & 15, lane2 = (q >> 4) & 31, t = (q >> 9) & 3, s = q >> 11;
        int half = lane2 >> 4;
        int kk = (j < 8 ? j : j + 8) + 8 * half;
        int k = s * 32 + kk;
        int n = t * 16 + (lane2 & 15);
        float v = (k < INDIM) ? W1[k * EH + n] : 0.0f;
        w1f[q] = (__bf16)v;
    }
    // --- W2 fragments ---
    for (int q = tidb; q < 2048; q += 256) {
        int j = q & 15, lane2 = (q >> 4) & 31, t = (q >> 9) & 1, s = q >> 10;
        int half = lane2 >> 4;
        int kk = (j < 8 ? j : j + 8) + 8 * half;
        int k = s * 32 + kk;
        int n = t * 16 + (lane2 & 15);
        w2f[q] = (__bf16)W2[k * EH2 + n];
    }
    // --- W3 fragments ---
    for (int q = tidb; q < 2048; q += 256) {
        int j = q & 15, lane2 = (q >> 4) & 31, t = q >> 9;
        int half = lane2 >> 4;
        int kk = (j < 8 ? j : j + 8) + 8 * half;
        int n = t * 16 + (lane2 & 15);
        w3f[q] = (__bf16)W3[kk * EOUT + n];
    }
    if (tidb < 64) b1f[tidb] = b1[tidb];
    else if (tidb < 96) b2f[tidb - 64] = b2[tidb - 64];
    else if (tidb < 160) b3f[tidb - 96] = b3[tidb - 96];
    __syncthreads();

    const int w    = tidb >> 5;
    const int lane = tidb & 31;
    const int m    = lane & 15;     // row (edge within tile) / N-column index
    const int half = lane >> 4;
    const int ET   = Ec / 16;       // tiles per batch
    const int numTiles = Bc * ET;
    const int hw1 = w * 16 * 72;
    const int hw2 = w * 16 * 40;

    for (int tid = blockIdx.x * 8 + w; tid < numTiles; tid += gridDim.x * 8) {
        const int b    = tid / ET;
        const int base = (tid % ET) * 16;
        const int e    = base + m;

        const int si = esrc[e];
        const int ti = etgt[e];
        if (!half) { idxS[w * 16 + m] = si; idxT[w * 16 + m] = ti; }

        // ---- gather: lanes 0-15 take src/tgt feats [0..7], 16-31 take [8..15]
        const float4* sp = (const float4*)(nf + ((size_t)b * Nc + si) * FNc + 8 * half);
        const float4* tp = (const float4*)(nf + ((size_t)b * Nc + ti) * FNc + 8 * half);
        float4 s0 = sp[0], s1 = sp[1], t0 = tp[0], t1 = tp[1];
        float  evw = ew[(size_t)b * Ec + e];

        v16bf a0;
        a0[0] = (__bf16)s0.x; a0[1] = (__bf16)s0.y; a0[2] = (__bf16)s0.z; a0[3] = (__bf16)s0.w;
        a0[4] = (__bf16)s1.x; a0[5] = (__bf16)s1.y; a0[6] = (__bf16)s1.z; a0[7] = (__bf16)s1.w;
        a0[8]  = (__bf16)t0.x; a0[9]  = (__bf16)t0.y; a0[10] = (__bf16)t0.z; a0[11] = (__bf16)t0.w;
        a0[12] = (__bf16)t1.x; a0[13] = (__bf16)t1.y; a0[14] = (__bf16)t1.z; a0[15] = (__bf16)t1.w;

        v16bf a1;
#pragma unroll
        for (int i = 0; i < 16; ++i) a1[i] = (__bf16)0.0f;
        a1[0] = (__bf16)(half ? 0.0f : evw);   // k==32 column = edge weight

        // ---- layer 1: 16x64, K=64 (2 k-steps) ----
#pragma unroll
        for (int t = 0; t < 4; ++t) {
            float bb = b1f[t * 16 + m];
            v8f c = {bb, bb, bb, bb, bb, bb, bb, bb};
            c = WMMA_BF16(a0, *(const v16bf*)&w1f[(t * 32 + lane) * 16], c);
            c = WMMA_BF16(a1, *(const v16bf*)&w1f[((4 + t) * 32 + lane) * 16], c);
#pragma unroll
            for (int r = 0; r < 8; ++r)
                h1s[hw1 + (r + 8 * half) * 72 + t * 16 + m] = (__bf16)fmaxf(c[r], 0.0f);
        }

        // ---- layer 2: 16x32, K=64 ----
        v16bf a20, a21;
#pragma unroll
        for (int s = 0; s < 2; ++s) {
            const __bf16* row = &h1s[hw1 + m * 72 + s * 32 + 8 * half];
            v8bf lo = *(const v8bf*)row;
            v8bf hi = *(const v8bf*)(row + 16);
            v16bf a = __builtin_shufflevector(lo, hi, 0,1,2,3,4,5,6,7,8,9,10,11,12,13,14,15);
            if (s == 0) a20 = a; else a21 = a;
        }
#pragma unroll
        for (int t = 0; t < 2; ++t) {
            float bb = b2f[t * 16 + m];
            v8f c = {bb, bb, bb, bb, bb, bb, bb, bb};
            c = WMMA_BF16(a20, *(const v16bf*)&w2f[(t * 32 + lane) * 16], c);
            c = WMMA_BF16(a21, *(const v16bf*)&w2f[((2 + t) * 32 + lane) * 16], c);
#pragma unroll
            for (int r = 0; r < 8; ++r)
                h2s[hw2 + (r + 8 * half) * 40 + t * 16 + m] = (__bf16)fmaxf(c[r], 0.0f);
        }

        // ---- layer 3: 16x64, K=32, sigmoid, scatter atomics ----
        {
            const __bf16* row = &h2s[hw2 + m * 40 + 8 * half];
            v8bf lo = *(const v8bf*)row;
            v8bf hi = *(const v8bf*)(row + 16);
            v16bf a3 = __builtin_shufflevector(lo, hi, 0,1,2,3,4,5,6,7,8,9,10,11,12,13,14,15);
#pragma unroll
            for (int t = 0; t < 4; ++t) {
                float bb = b3f[t * 16 + m];
                v8f c = {bb, bb, bb, bb, bb, bb, bb, bb};
                c = WMMA_BF16(a3, *(const v16bf*)&w3f[(t * 32 + lane) * 16], c);
#pragma unroll
                for (int r = 0; r < 8; ++r) {
                    float val = 1.0f / (1.0f + __expf(-c[r]));
                    int M = r + 8 * half;           // edge row within tile
                    int n = t * 16 + m;             // output feature
                    int tg = idxT[w * 16 + M];
                    int sg = idxS[w * 16 + M];
                    float* pt = agg + ((size_t)b * Nc + tg) * 64 + n;
                    float* ps = agg + ((size_t)b * Nc + sg) * 64 + n;
                    __hip_atomic_fetch_add(pt,  val, __ATOMIC_RELAXED, __HIP_MEMORY_SCOPE_AGENT);
                    __hip_atomic_fetch_add(ps, -val, __ATOMIC_RELAXED, __HIP_MEMORY_SCOPE_AGENT);
                }
            }
        }
    }
}

// ---------------------------------------------------------------- node MLP
__global__ __launch_bounds__(256) void egnn_node_kernel(
    const float* __restrict__ agg,     // (B,N,64)
    const float* __restrict__ lp,      // (B,N,1)
    const float* __restrict__ W4, const float* __restrict__ b4,   // (65,12),(12)
    const float* __restrict__ W5, const float* __restrict__ b5,   // (12,1),(1)
    float* __restrict__ out)           // (B,N,1)
{
    int idx = blockIdx.x * blockDim.x + threadIdx.x;
    if (idx >= Bc * Nc) return;
    const float* a = agg + (size_t)idx * 64;
    float acc[NH];
#pragma unroll
    for (int j = 0; j < NH; ++j) acc[j] = b4[j];
#pragma unroll 4
    for (int i = 0; i < 64; ++i) {
        float av = a[i];
#pragma unroll
        for (int j = 0; j < NH; ++j) acc[j] = fmaf(av, W4[i * NH + j], acc[j]);
    }
    float l = lp[idx];
#pragma unroll
    for (int j = 0; j < NH; ++j) acc[j] = fmaf(l, W4[64 * NH + j], acc[j]);
    float o = b5[0];
#pragma unroll
    for (int j = 0; j < NH; ++j) o = fmaf(fmaxf(acc[j], 0.0f), W5[j], o);
    out[idx] = 1.0f / (1.0f + __expf(-o));
}

// ---------------------------------------------------------------- launcher
extern "C" void kernel_launch(void* const* d_in, const int* in_sizes, int n_in,
                              void* d_out, int out_size, void* d_ws, size_t ws_size,
                              hipStream_t stream) {
    const float* nf   = (const float*)d_in[0];
    const float* ew   = (const float*)d_in[1];
    const float* lp   = (const float*)d_in[2];
    const int*   esrc = (const int*)d_in[3];
    const int*   etgt = (const int*)d_in[4];
    const float* W1 = (const float*)d_in[5];
    const float* b1 = (const float*)d_in[6];
    const float* W2 = (const float*)d_in[7];
    const float* b2 = (const float*)d_in[8];
    const float* W3 = (const float*)d_in[9];
    const float* b3 = (const float*)d_in[10];
    const float* W4 = (const float*)d_in[11];
    const float* b4 = (const float*)d_in[12];
    const float* W5 = (const float*)d_in[13];
    const float* b5 = (const float*)d_in[14];
    float* out = (float*)d_out;

    float* agg = (float*)d_ws;                       // B*N*64 floats = 10.24 MB
    const int aggN = Bc * Nc * 64;

    egnn_zero_kernel<<<2500, 256, 0, stream>>>((float4*)agg, aggN / 4);
    egnn_edge_kernel<<<2500, 256, 0, stream>>>(nf, ew, esrc, etgt,
                                               W1, b1, W2, b2, W3, b3, agg);
    egnn_node_kernel<<<(Bc * Nc + 255) / 256, 256, 0, stream>>>(agg, lp, W4, b4, W5, b5, out);
    (void)in_sizes; (void)n_in; (void)out_size; (void)ws_size;
}
